// SelfAttention_55198919688805
// MI455X (gfx1250) — matrix operand order repaired and checked by reading, not compiled, not run on hardware
//
#include <hip/hip_runtime.h>

// ---------------------------------------------------------------------------
// Types for CDNA5 WMMA (wave32): 16x16x32 bf16 -> f32, and TDM descriptors
// ---------------------------------------------------------------------------
typedef __attribute__((ext_vector_type(8)))  __bf16 bf16x8;
typedef __attribute__((ext_vector_type(16))) __bf16 bf16x16;
typedef __attribute__((ext_vector_type(8)))  float  floatx8;
typedef __attribute__((ext_vector_type(4)))  unsigned int uint32x4;
typedef __attribute__((ext_vector_type(8)))  int  int32x8;
typedef __attribute__((ext_vector_type(4)))  int  int32x4;

__device__ __forceinline__ unsigned short f32_to_bf16(float f) {
  unsigned int u = __float_as_uint(f);
  u += 0x7fffu + ((u >> 16) & 1u);   // round-to-nearest-even
  return (unsigned short)(u >> 16);
}

// Generic LDS pointer -> LDS byte offset (flat LDS address = addr[31:0])
__device__ __forceinline__ unsigned int lds_offset_of(const void* p) {
  return (unsigned int)(unsigned long long)(uintptr_t)p;
}

// ---------------------------------------------------------------------------
// f32 -> bf16 bulk convert (vectorized: float4 in, ushort4 out)
// ---------------------------------------------------------------------------
__global__ __launch_bounds__(256)
void convert_f32_to_bf16_kernel(const float* __restrict__ in,
                                unsigned short* __restrict__ out, long n4) {
  long i = (long)blockIdx.x * blockDim.x + threadIdx.x;
  if (i >= n4) return;
  float4 f = ((const float4*)in)[i];
  ushort4 o;
  o.x = f32_to_bf16(f.x);
  o.y = f32_to_bf16(f.y);
  o.z = f32_to_bf16(f.z);
  o.w = f32_to_bf16(f.w);
  ((ushort4*)out)[i] = o;
}

// ---------------------------------------------------------------------------
// W (1024x1024 f32, row-major [k][n]) -> Wt (bf16, row-major [n][k])
// ---------------------------------------------------------------------------
__global__ __launch_bounds__(256)
void transpose_w_kernel(const float* __restrict__ W,
                        unsigned short* __restrict__ Wt) {
  const int k = blockIdx.x * 32 + (threadIdx.x & 31);
  const int n0 = blockIdx.y * 32 + (threadIdx.x >> 5);
#pragma unroll
  for (int nn = 0; nn < 32; nn += 8) {
    const int n = n0 + nn;
    Wt[(long)n * 1024 + k] = f32_to_bf16(W[(long)k * 1024 + n]);
  }
}

// ---------------------------------------------------------------------------
// TDM: async-load a 128-row x 32-bf16 tile (row stride = stride_elems) from
// global into LDS at lds_byte_off, with hardware padding of 4 DWORDs (16B)
// after every 16 DWORDs (64B = one 32-elem row) -> LDS pitch 40 ushorts,
// matching the bank-conflict-free layout used by the compute waves.
// Descriptor bit-packing per CDNA5 ISA.
// This toolchain exposes the 6-arg builtin:
//   (uint32x4 g0, int32x8 g1, int32x4 g2, int32x4 g3, int32x8 extra, i32 cpol)
// groups 2/3 and the extra group are unused for a 2D tile -> zeros.
// ---------------------------------------------------------------------------
#define LDS_PITCH 40

#if __has_builtin(__builtin_amdgcn_tensor_load_to_lds)
#define USE_TDM 1
__device__ __forceinline__ void tdm_load_tile_128x32(unsigned int lds_byte_off,
                                                     const unsigned short* gsrc,
                                                     unsigned int stride_elems) {
  const unsigned long long ga = (unsigned long long)(uintptr_t)gsrc;
  // Group 0: count=1 | lds_addr | global_addr[56:0] | type=2
  uint32x4 g0 = {1u, lds_byte_off, (unsigned int)ga,
                 (unsigned int)((ga >> 32) & 0x01ffffffu) | (2u << 30)};
  const unsigned int td0 = 32u, td1 = 128u;   // tensor dims == tile dims here
  int32x8 g1;
  g1[0] = (int)((1u << 16)      // data_size = 1 -> 2 bytes
              | (1u << 20)      // pad_enable
              | (3u << 22)      // pad_interval: 16 DWORDs (64B) per pad
              | (3u << 25));    // pad_amount: 4 DWORDs (16B)
  g1[1] = (int)((td0 & 0xffffu) << 16);              // tensor_dim0[15:0]
  g1[2] = (int)((td0 >> 16) | ((td1 & 0xffffu) << 16)); // dim0 hi | dim1 lo
  g1[3] = (int)((td1 >> 16) | (32u << 16));          // dim1 hi | tile_dim0=32
  g1[4] = (int)(128u);                               // tile_dim1=128, tile_dim2=0
  g1[5] = (int)stride_elems;                         // tensor_dim0_stride[31:0]
  g1[6] = 0;                                         // stride0 hi | stride1 lo
  g1[7] = 0;
  int32x4 z4 = {0, 0, 0, 0};                         // groups 2/3 unused (2D)
  int32x8 z8 = {0, 0, 0, 0, 0, 0, 0, 0};
  __builtin_amdgcn_tensor_load_to_lds(g0, g1, z4, z4, z8, 0);
}
#endif

// ---------------------------------------------------------------------------
// Tiled bf16 GEMM:  C[z][m][n] = sum_k A[z][m][k] * Bt[z][n][k]
// Block tile 128x128, BK=32, 256 threads = 8 waves (2M x 4N), each wave
// computes 4x2 WMMA 16x16 tiles. Tiles staged to LDS via TDM (double-buffered,
// one-stage software pipeline) or via global->ds fallback.
// ---------------------------------------------------------------------------
template <bool OUT_BF16>
__global__ __launch_bounds__(256)
void gemm_bt_kernel(const unsigned short* __restrict__ A, int lda, long strideA,
                    const unsigned short* __restrict__ Bt, int ldb, long strideB,
                    void* __restrict__ Cout, int ldc, long strideC, int Kdim) {
  __shared__ __align__(16) unsigned short As[2][128 * LDS_PITCH];
  __shared__ __align__(16) unsigned short Bs[2][128 * LDS_PITCH];

  const int tid   = threadIdx.x;
  const int lane  = tid & 31;
  const int w     = tid >> 5;
  const int waveM = w & 1;   // 0..1
  const int waveN = w >> 1;  // 0..3
  const int bm = blockIdx.y * 128;
  const int bn = blockIdx.x * 128;
  const int z  = blockIdx.z;

  A  += (long)z * strideA + (long)bm * lda;
  Bt += (long)z * strideB + (long)bn * ldb;

  floatx8 acc[4][2];
#pragma unroll
  for (int mi = 0; mi < 4; ++mi)
#pragma unroll
    for (int ni = 0; ni < 2; ++ni)
#pragma unroll
      for (int r = 0; r < 8; ++r) acc[mi][ni][r] = 0.0f;

  const int ar = lane & 15;          // row within 16
  const int ak = (lane >> 4) * 8;    // A frag: K base 0/8 (pairs with +16)
  const int bk = (lane >> 4) * 8;    // B frag: K base 0/8 (pairs with +16)

  const int nsteps = Kdim / 32;

#if defined(USE_TDM)
  // Wave 0 drives the Tensor Data Mover; TENSORcnt tracks completion.
  if (tid < 32) {
    tdm_load_tile_128x32(lds_offset_of(&As[0][0]), A, (unsigned)lda);
    tdm_load_tile_128x32(lds_offset_of(&Bs[0][0]), Bt, (unsigned)ldb);
  }
#endif

  for (int s = 0; s < nsteps; ++s) {
#if defined(USE_TDM)
    const int buf = s & 1;
    if (tid < 32) {
      if (s + 1 < nsteps) {
        const long kn = (long)(s + 1) * 32;
        tdm_load_tile_128x32(lds_offset_of(&As[buf ^ 1][0]), A + kn,
                             (unsigned)lda);
        tdm_load_tile_128x32(lds_offset_of(&Bs[buf ^ 1][0]), Bt + kn,
                             (unsigned)ldb);
        __builtin_amdgcn_s_wait_tensorcnt(2);  // current stage has landed
      } else {
        __builtin_amdgcn_s_wait_tensorcnt(0);
      }
    }
#else
    const int buf = 0;
    const int k0 = s * 32;
#pragma unroll
    for (int i = 0; i < 2; ++i) {
      const int c   = tid + i * 256;
      const int row = c >> 2;
      const int off = (c & 3) * 8;
      *(uint4*)(&As[0][row * LDS_PITCH + off]) =
          *(const uint4*)(&A[(long)row * lda + k0 + off]);
      *(uint4*)(&Bs[0][row * LDS_PITCH + off]) =
          *(const uint4*)(&Bt[(long)row * ldb + k0 + off]);
    }
#endif
    __syncthreads();

    const unsigned short* as = &As[buf][0];
    const unsigned short* bs = &Bs[buf][0];

    // A frag: lane<16 -> M=lane, K={0..7,16..23}; lane>=16 -> K={8..15,24..31}
    bf16x16 afrag[4];
#pragma unroll
    for (int mi = 0; mi < 4; ++mi) {
      const int row = waveM * 64 + mi * 16 + ar;
      union { bf16x16 v; bf16x8 h[2]; } u;
      u.h[0] = *(const bf16x8*)(&as[row * LDS_PITCH + ak]);
      u.h[1] = *(const bf16x8*)(&as[row * LDS_PITCH + ak + 16]);
      afrag[mi] = u.v;
    }
    // B frag: lane<16 -> N=lane, K={0..7,16..23}; lane>=16 -> K={8..15,24..31}
    bf16x16 bfrag[2];
#pragma unroll
    for (int ni = 0; ni < 2; ++ni) {
      const int row = waveN * 32 + ni * 16 + ar;
      union { bf16x16 v; bf16x8 h[2]; } u;
      u.h[0] = *(const bf16x8*)(&bs[row * LDS_PITCH + bk]);
      u.h[1] = *(const bf16x8*)(&bs[row * LDS_PITCH + bk + 16]);
      bfrag[ni] = u.v;
    }

#pragma unroll
    for (int mi = 0; mi < 4; ++mi)
#pragma unroll
      for (int ni = 0; ni < 2; ++ni)
        acc[mi][ni] = __builtin_amdgcn_wmma_f32_16x16x32_bf16(
            false, afrag[mi], false, bfrag[ni], (short)0, acc[mi][ni], false,
            false);
    __syncthreads();  // protects LDS buffer reuse in the next stages
  }

  // Store: VGPR r of lane l -> row = r + (l>=16 ? 8 : 0), col = l&15
  const int cn = lane & 15;
  const int mo = (lane >> 4) * 8;
#pragma unroll
  for (int mi = 0; mi < 4; ++mi)
#pragma unroll
    for (int ni = 0; ni < 2; ++ni) {
      const int rowbase = bm + waveM * 64 + mi * 16 + mo;
      const int col     = bn + waveN * 32 + ni * 16 + cn;
#pragma unroll
      for (int r = 0; r < 8; ++r) {
        const long idx = (long)z * strideC + (long)(rowbase + r) * ldc + col;
        const float v = acc[mi][ni][r];
        if (OUT_BF16)
          ((unsigned short*)Cout)[idx] = f32_to_bf16(v);
        else
          ((float*)Cout)[idx] = v;
      }
    }
}

// ---------------------------------------------------------------------------
// Masked-score output: out0[b][i][j] = (i<m_b && j<m_b) ? S[b][i][j] : 1000
// ---------------------------------------------------------------------------
__global__ __launch_bounds__(256)
void mask_kernel(const float* __restrict__ S, const int* __restrict__ mask,
                 float* __restrict__ out) {
  const int b = blockIdx.x;
  const int m = mask[b];
  const float* Sb = S + (long)b * 512 * 512;
  float* ob = out + (long)b * 64 * 64;
  for (int t = threadIdx.x; t < 64 * 64; t += blockDim.x) {
    const int i = t >> 6, j = t & 63;
    const bool valid = (i < m) && (j < m);
    ob[t] = valid ? Sb[i * 512 + j] : 1000.0f;
  }
}

// ---------------------------------------------------------------------------
// Softmax over 512-wide rows: one wave32 per row, 16 elems/lane.
// ---------------------------------------------------------------------------
__global__ __launch_bounds__(256)
void softmax_kernel(const float* __restrict__ S, float* __restrict__ out) {
  const long row  = ((long)blockIdx.x * blockDim.x + threadIdx.x) >> 5;
  const int  lane = threadIdx.x & 31;
  const float* src = S + row * 512;
  float v[16];
  float mx = -3.4e38f;
#pragma unroll
  for (int i = 0; i < 16; ++i) {
    v[i] = src[lane + i * 32];
    mx = fmaxf(mx, v[i]);
  }
#pragma unroll
  for (int o = 16; o > 0; o >>= 1) mx = fmaxf(mx, __shfl_xor(mx, o, 32));
  float sum = 0.0f;
#pragma unroll
  for (int i = 0; i < 16; ++i) {
    v[i] = __expf(v[i] - mx);
    sum += v[i];
  }
#pragma unroll
  for (int o = 16; o > 0; o >>= 1) sum += __shfl_xor(sum, o, 32);
  const float inv = 1.0f / sum;
  float* dst = out + row * 512;
#pragma unroll
  for (int i = 0; i < 16; ++i) dst[lane + i * 32] = v[i] * inv;
}

// ---------------------------------------------------------------------------
// Host-side orchestration
// ---------------------------------------------------------------------------
extern "C" void kernel_launch(void* const* d_in, const int* in_sizes, int n_in,
                              void* d_out, int out_size, void* d_ws,
                              size_t ws_size, hipStream_t stream) {
  (void)in_sizes; (void)n_in; (void)out_size; (void)ws_size;

  const int B = 64, S = 512, D = 1024;
  const long MS = (long)B * S;  // 32768 rows

  const float* X    = (const float*)d_in[0];
  const int*   mask = (const int*)d_in[1];
  const float* Wq   = (const float*)d_in[2];
  const float* Wk   = (const float*)d_in[3];
  // d_in[4] (Wv) is dead in the reference.

  unsigned short* Xb  = (unsigned short*)d_ws;       // 32768x1024 bf16
  unsigned short* Wqt = Xb  + MS * D;                // 1024x1024  bf16 (T)
  unsigned short* Wkt = Wqt + (long)D * D;           // 1024x1024  bf16 (T)
  unsigned short* Qb  = Wkt + (long)D * D;           // 32768x1024 bf16
  unsigned short* Kb  = Qb  + MS * D;                // 32768x1024 bf16
  float*          Sr  = (float*)(Kb + MS * D);       // 64x512x512 f32

  float* out_mask = (float*)d_out;                   // 64*64*64
  float* out_sm   = out_mask + (long)B * 64 * 64;    // 64*512*512

  {
    const long n4 = MS * D / 4;
    convert_f32_to_bf16_kernel<<<(unsigned)(n4 / 256), 256, 0, stream>>>(X, Xb, n4);
  }
  transpose_w_kernel<<<dim3(32, 32), 256, 0, stream>>>(Wq, Wqt);
  transpose_w_kernel<<<dim3(32, 32), 256, 0, stream>>>(Wk, Wkt);

  gemm_bt_kernel<true><<<dim3(D / 128, MS / 128, 1), 256, 0, stream>>>(
      Xb, D, 0L, Wqt, D, 0L, Qb, D, 0L, D);
  gemm_bt_kernel<true><<<dim3(D / 128, MS / 128, 1), 256, 0, stream>>>(
      Xb, D, 0L, Wkt, D, 0L, Kb, D, 0L, D);

  gemm_bt_kernel<false><<<dim3(S / 128, S / 128, B), 256, 0, stream>>>(
      Qb, D, (long)S * D, Kb, D, (long)S * D, Sr, S, (long)S * S, D);

  mask_kernel<<<B, 256, 0, stream>>>(Sr, mask, out_mask);
  softmax_kernel<<<(unsigned)(B * S / 8), 256, 0, stream>>>(Sr, out_sm);
}
